// DataParalleledHAN_73237782332045
// MI455X (gfx1250) — compile-verified, matrix operand order checked by reading, not run on hardware
//
#include <hip/hip_runtime.h>
#include <math.h>

#define NB 10000
#define CH 128
#define NE 160000
#define NH 8
#define MT 625   // NB / 16

typedef unsigned short u16;
typedef __attribute__((ext_vector_type(16))) __bf16 v16bf;
typedef __attribute__((ext_vector_type(8)))  float  v8f;

union AFrag { v16bf v; uint4 q[2]; };

__device__ __forceinline__ u16 f2bf(float f) {
  unsigned u = __float_as_uint(f);
  u += 0x7FFFu + ((u >> 16) & 1u);           // round-to-nearest-even
  return (u16)(u >> 16);
}
// monotone float<->uint encoding so unsigned atomicMax orders like float max
__device__ __forceinline__ unsigned fenc(float f) {
  unsigned u = __float_as_uint(f);
  return (u & 0x80000000u) ? ~u : (u | 0x80000000u);
}
__device__ __forceinline__ float fdec(unsigned e) {
  unsigned u = (e & 0x80000000u) ? (e & 0x7FFFFFFFu) : ~e;
  return __uint_as_float(u);
}

// ---------------- utility ----------------
__global__ void k_zero(unsigned* p, long n) {
  long i = (long)blockIdx.x * blockDim.x + threadIdx.x;
  long st = (long)gridDim.x * blockDim.x;
  for (; i < n; i += st) p[i] = 0u;
}

__global__ void k_relu(float* p, long n) {
  long i = (long)blockIdx.x * blockDim.x + threadIdx.x;
  long st = (long)gridDim.x * blockDim.x;
  for (; i < n; i += st) p[i] = fmaxf(p[i], 0.f);
}

// ---------------- bf16 fragment packing (ISA 7.12.2 layouts) ----------------
// A 16x32 bf16 frag: lane L<16 -> row L, K {0..7,16..23}; lane L>=16 -> K {8..15,24..31}
__device__ __forceinline__ int kpat(int j, int half) {
  return (j < 8) ? (half * 8 + j) : (16 + half * 8 + (j - 8));
}

__global__ void k_pack_a(const float* __restrict__ x, u16* __restrict__ out, int KT, int K) {
  int idx = blockIdx.x * blockDim.x + threadIdx.x;
  if (idx >= MT * KT * 512) return;
  int j    = idx & 15;
  int lane = (idx >> 4) & 31;
  int frag = idx >> 9;
  int kt = frag % KT, mt = frag / KT;
  int k   = kt * 32 + kpat(j, lane >> 4);
  int row = mt * 16 + (lane & 15);
  out[idx] = f2bf(x[(long)row * K + k]);
}

// B frag (32x16, column-major per tile), W is row-major [K, 128]
__global__ void k_pack_b(const float* __restrict__ w, u16* __restrict__ out, int KT) {
  int idx = blockIdx.x * blockDim.x + threadIdx.x;
  if (idx >= 8 * KT * 512) return;
  int j    = idx & 15;
  int lane = (idx >> 4) & 31;
  int frag = idx >> 9;
  int kt = frag % KT, nt = frag / KT;
  int k   = kt * 32 + kpat(j, lane >> 4);
  int col = nt * 16 + (lane & 15);
  out[idx] = f2bf(w[(long)k * CH + col]);
}

// ---------------- WMMA GEMM: [10000 x K] * [K x 128] + bias ----------------
__global__ void __launch_bounds__(256) k_gemm(const u16* __restrict__ A,
                                              const u16* __restrict__ B,
                                              const float* __restrict__ bias,
                                              float* __restrict__ out, int KT) {
  int wave = threadIdx.x >> 5, lane = threadIdx.x & 31;
  int mt = blockIdx.x * 8 + wave;
  if (mt >= MT) return;                       // wave-uniform: EXEC all-1 inside
  int col0 = lane & 15;
  v8f acc[8];
  #pragma unroll
  for (int nt = 0; nt < 8; ++nt) {
    float bv = bias[nt * 16 + col0];
    #pragma unroll
    for (int r = 0; r < 8; ++r) acc[nt][r] = bv;
  }
  for (int kt = 0; kt < KT; ++kt) {
    AFrag a;
    const uint4* ap = (const uint4*)(A + ((long)(mt * KT + kt) * 32 + lane) * 16);
    a.q[0] = ap[0]; a.q[1] = ap[1];
    if (kt + 1 < KT)
      __builtin_prefetch(A + ((long)(mt * KT + kt + 1) * 32 + lane) * 16, 0, 0);
    #pragma unroll
    for (int nt = 0; nt < 8; ++nt) {
      AFrag b;
      const uint4* bp = (const uint4*)(B + ((long)(nt * KT + kt) * 32 + lane) * 16);
      b.q[0] = bp[0]; b.q[1] = bp[1];
      acc[nt] = __builtin_amdgcn_wmma_f32_16x16x32_bf16(false, a.v, false, b.v,
                                                        (short)0, acc[nt], false, false);
    }
  }
  int rowbase = mt * 16 + (lane >> 4) * 8;
  #pragma unroll
  for (int nt = 0; nt < 8; ++nt)
    #pragma unroll
    for (int r = 0; r < 8; ++r)
      out[(long)(rowbase + r) * CH + nt * 16 + col0] = acc[nt][r];
}

// WMMA GEMM with tanh + q-dot epilogue, block-reduced into a scalar score
__global__ void __launch_bounds__(256) k_gemm_score(const u16* __restrict__ A,
                                                    const u16* __restrict__ B,
                                                    const float* __restrict__ bias,
                                                    const float* __restrict__ q,
                                                    float* __restrict__ score, int KT) {
  int wave = threadIdx.x >> 5, lane = threadIdx.x & 31;
  int mt = blockIdx.x * 8 + wave;
  int col0 = lane & 15;
  float part = 0.f;
  if (mt < MT) {
    v8f acc[8];
    #pragma unroll
    for (int nt = 0; nt < 8; ++nt) {
      float bv = bias[nt * 16 + col0];
      #pragma unroll
      for (int r = 0; r < 8; ++r) acc[nt][r] = bv;
    }
    for (int kt = 0; kt < KT; ++kt) {
      AFrag a;
      const uint4* ap = (const uint4*)(A + ((long)(mt * KT + kt) * 32 + lane) * 16);
      a.q[0] = ap[0]; a.q[1] = ap[1];
      #pragma unroll
      for (int nt = 0; nt < 8; ++nt) {
        AFrag b;
        const uint4* bp = (const uint4*)(B + ((long)(nt * KT + kt) * 32 + lane) * 16);
        b.q[0] = bp[0]; b.q[1] = bp[1];
        acc[nt] = __builtin_amdgcn_wmma_f32_16x16x32_bf16(false, a.v, false, b.v,
                                                          (short)0, acc[nt], false, false);
      }
    }
    #pragma unroll
    for (int nt = 0; nt < 8; ++nt) {
      float qv = q[nt * 16 + col0];
      #pragma unroll
      for (int r = 0; r < 8; ++r) part += qv * tanhf(acc[nt][r]);
    }
  }
  __shared__ float red[256];
  red[threadIdx.x] = part;
  __syncthreads();
  for (int s = 128; s > 0; s >>= 1) {
    if ((int)threadIdx.x < s) red[threadIdx.x] += red[threadIdx.x + s];
    __syncthreads();
  }
  if (threadIdx.x == 0) atomicAdd(score, red[0]);
}

// ---------------- attention scalars: a[n,h] = sum_d h[n,h*16+d]*att[h,d] ----------------
__global__ void k_headdot(const float* __restrict__ h, const float* __restrict__ att,
                          float* __restrict__ out) {
  int idx = blockIdx.x * blockDim.x + threadIdx.x;
  if (idx >= NB * NH) return;
  int n = idx >> 3, hh = idx & 7;
  const float* hp = h + (long)n * CH + hh * 16;
  const float* ap = att + hh * 16;
  float s = 0.f;
  #pragma unroll
  for (int d = 0; d < 16; ++d) s += hp[d] * ap[d];
  out[idx] = s;
}

// ---------------- edge softmax passes ----------------
__global__ void k_edge_max(const int* __restrict__ ei, const float* __restrict__ asrc,
                           const float* __restrict__ adst, float* __restrict__ araw,
                           unsigned* __restrict__ mbuf) {
  int idx = blockIdx.x * blockDim.x + threadIdx.x;
  if (idx >= NE * NH) return;
  int e = idx >> 3, hh = idx & 7;
  int s = ei[e], d = ei[NE + e];
  float a = asrc[s * NH + hh] + adst[d * NH + hh];
  a = (a > 0.f) ? a : 0.2f * a;                 // leaky_relu(0.2)
  araw[idx] = a;
  atomicMax(&mbuf[d * NH + hh], fenc(a));
}

__global__ void k_edge_exp(const int* __restrict__ ei, float* __restrict__ araw,
                           const unsigned* __restrict__ mbuf, float* __restrict__ ssum) {
  int idx = blockIdx.x * blockDim.x + threadIdx.x;
  if (idx >= NE * NH) return;
  int e = idx >> 3, hh = idx & 7;
  int d = ei[NE + e];
  float ex = __expf(araw[idx] - fdec(mbuf[d * NH + hh]));
  araw[idx] = ex;
  atomicAdd(&ssum[d * NH + hh], ex);
}

__global__ void __launch_bounds__(256) k_edge_scatter(const int* __restrict__ ei,
                                                      const float* __restrict__ ex,
                                                      const float* __restrict__ ssum,
                                                      const float* __restrict__ hs,
                                                      float* __restrict__ agg) {
  int e = blockIdx.x * 2 + (threadIdx.x >> 7);
  int c = threadIdx.x & 127;
  if (e >= NE) return;
  int s = ei[e], d = ei[NE + e];
  int hh = c >> 4;
  float w = ex[e * NH + hh] / (ssum[d * NH + hh] + 1e-16f);
  atomicAdd(&agg[(long)d * CH + c], hs[(long)s * CH + c] * w);
}

// ---------------- semantic softmax combine ----------------
__global__ void k_combine(const float* __restrict__ a0, const float* __restrict__ a1,
                          const float* __restrict__ sc0, const float* __restrict__ sc1,
                          int R, float* __restrict__ out) {
  int i = blockIdx.x * blockDim.x + threadIdx.x;
  if (i >= NB * CH) return;
  float w0 = 1.f, w1 = 0.f;
  if (R == 2) {
    float s0 = sc0[0] * (1.f / NB), s1 = sc1[0] * (1.f / NB);
    float m = fmaxf(s0, s1);
    float e0 = __expf(s0 - m), e1 = __expf(s1 - m);
    float inv = 1.f / (e0 + e1);
    w0 = e0 * inv; w1 = e1 * inv;
  }
  float v = w0 * a0[i];
  if (R == 2) v += w1 * a1[i];
  out[i] = v;
}

// ---------------- host orchestration ----------------
extern "C" void kernel_launch(void* const* d_in, const int* in_sizes, int n_in,
                              void* d_out, int out_size, void* d_ws, size_t ws_size,
                              hipStream_t stream) {
  (void)in_sizes; (void)n_in; (void)out_size; (void)ws_size;
  const float* xin[3] = {(const float*)d_in[0], (const float*)d_in[1], (const float*)d_in[2]};
  const int*   ed[4]  = {(const int*)d_in[3], (const int*)d_in[4],
                         (const int*)d_in[5], (const int*)d_in[6]};

  char* ws = (char*)d_ws;
  size_t off = 0;
  auto take = [&](size_t bytes) -> char* {
    char* p = ws + off;
    off += (bytes + 255) & ~(size_t)255;
    return p;
  };
  float* Hb[3];  for (int t = 0; t < 3; ++t) Hb[t] = (float*)take((size_t)NB * CH * 4);
  float* Xa[3];  for (int t = 0; t < 3; ++t) Xa[t] = (float*)take((size_t)NB * CH * 4);
  u16* ApX[3];   for (int t = 0; t < 3; ++t) ApX[t] = (u16*)take((size_t)NB * CH * 2);
  u16* ApR = (u16*)take((size_t)NB * CH * 2);
  u16* Wp[3];    for (int t = 0; t < 3; ++t) Wp[t] = (u16*)take((size_t)CH * CH * 2);
  u16* WpS = (u16*)take((size_t)CH * CH * 2);
  float* asrc = (float*)take((size_t)NB * NH * 4);
  float* adst = (float*)take((size_t)NB * NH * 4);
  unsigned* mb = (unsigned*)take((size_t)NB * NH * 4);
  float* ss = (float*)take((size_t)NB * NH * 4);
  float* exb = (float*)take((size_t)NE * NH * 4);
  float* agg[4]; for (int i = 0; i < 4; ++i) agg[i] = (float*)take((size_t)NB * CH * 4);
  float* scb = (float*)take(256);

  const int esrc[4] = {0, 1, 2, 1}, edst[4] = {1, 0, 1, 2};

  for (int L = 0; L < 8; ++L) {
    int base = 7 + L * 17;      // proj: W_A,b_A,W_B,b_B,W_C,b_C; att_src[4]; att_dst[4]; Wk,bk,q
    int K  = (L == 0) ? 64 : 128;
    int KT = K / 32;
    const float* xc[3];
    for (int t = 0; t < 3; ++t) xc[t] = (L == 0) ? xin[t] : Xa[t];

    // projections
    for (int t = 0; t < 3; ++t) {
      const float* W = (const float*)d_in[base + 2 * t];
      const float* b = (const float*)d_in[base + 2 * t + 1];
      k_pack_b<<<(8 * KT * 512 + 255) / 256, 256, 0, stream>>>(W, Wp[t], KT);
      k_pack_a<<<(MT * KT * 512 + 255) / 256, 256, 0, stream>>>(xc[t], ApX[t], KT, K);
      k_gemm<<<(MT + 7) / 8, 256, 0, stream>>>(ApX[t], Wp[t], b, Hb[t], KT);
    }
    k_pack_b<<<(8 * 4 * 512 + 255) / 256, 256, 0, stream>>>(
        (const float*)d_in[base + 14], WpS, 4);

    // edge attention + aggregation
    for (int i = 0; i < 4; ++i) {
      int st = esrc[i], dt = edst[i];
      k_headdot<<<(NB * NH + 255) / 256, 256, 0, stream>>>(
          Hb[st], (const float*)d_in[base + 6 + i], asrc);
      k_headdot<<<(NB * NH + 255) / 256, 256, 0, stream>>>(
          Hb[dt], (const float*)d_in[base + 10 + i], adst);
      k_zero<<<256, 256, 0, stream>>>(mb, (long)NB * NH);
      k_zero<<<256, 256, 0, stream>>>((unsigned*)ss, (long)NB * NH);
      k_zero<<<512, 256, 0, stream>>>((unsigned*)agg[i], (long)NB * CH);
      k_edge_max<<<(NE * NH + 255) / 256, 256, 0, stream>>>(ed[i], asrc, adst, exb, mb);
      k_edge_exp<<<(NE * NH + 255) / 256, 256, 0, stream>>>(ed[i], exb, mb, ss);
      k_edge_scatter<<<NE / 2, 256, 0, stream>>>(ed[i], exb, ss, Hb[st], agg[i]);
      k_relu<<<512, 256, 0, stream>>>(agg[i], (long)NB * CH);
    }

    // semantic scores (q . tanh(agg @ Wk + bk), summed; mean applied in combine)
    k_zero<<<1, 32, 0, stream>>>((unsigned*)scb, 4);
    const float* bk = (const float*)d_in[base + 15];
    const float* q  = (const float*)d_in[base + 16];
    for (int i = 0; i < 4; ++i) {
      k_pack_a<<<(MT * 4 * 512 + 255) / 256, 256, 0, stream>>>(agg[i], ApR, 4, 128);
      k_gemm_score<<<(MT + 7) / 8, 256, 0, stream>>>(ApR, WpS, bk, q, scb + i, 4);
    }

    // combine per destination node type: A <- rel{1}, B <- rel{0,2}, C <- rel{3}
    float* outp[3];
    for (int t = 0; t < 3; ++t)
      outp[t] = (L == 7) ? ((float*)d_out + (long)t * NB * CH) : Xa[t];
    int blks = (NB * CH + 255) / 256;
    k_combine<<<blks, 256, 0, stream>>>(agg[1], agg[1], scb + 1, scb + 1, 1, outp[0]);
    k_combine<<<blks, 256, 0, stream>>>(agg[0], agg[2], scb + 0, scb + 2, 2, outp[1]);
    k_combine<<<blks, 256, 0, stream>>>(agg[3], agg[3], scb + 3, scb + 3, 1, outp[2]);
  }
}
